// WMHCA3D_4715874091871
// MI455X (gfx1250) — matrix operand order repaired and checked by reading, not compiled, not run on hardware
//
#include <hip/hip_runtime.h>
#include <hip/hip_bf16.h>

typedef __attribute__((ext_vector_type(16))) _Float16 v16h;
typedef __attribute__((ext_vector_type(8)))  _Float16 v8h;
typedef __attribute__((ext_vector_type(8)))  float    v8f;

#define C_CH   128
#define DVOL   32
#define HVOL   64
#define WVOL   64
#define VOXELS (DVOL*HVOL*WVOL)   /* 131072 */
#define DSTRIDE (HVOL*WVOL)       /* 4096 */
#define NTAPS  27
#define WSLICE (C_CH*C_CH)        /* 16384 per tap */
#define LDA    136                /* padded f16 row stride (16B-aligned rows) */

// ---------------- WMMA fragment loaders (ISA 7.12.2 lane layouts) ------------
// A (16x32 f16): lane<16 -> row=lane, K {0..7, 16..23}; lane>=16 -> row=lane-16, K {8..15, 24..31}
__device__ __forceinline__ v16h frag_a(const _Float16* base, int row, int kbase, int lane) {
  const _Float16* p = base + (row + (lane & 15)) * LDA + kbase + ((lane >> 4) << 3);
  v8h lo = *(const v8h*)p;
  v8h hi = *(const v8h*)(p + 16);
  return __builtin_shufflevector(lo, hi, 0,1,2,3,4,5,6,7,8,9,10,11,12,13,14,15);
}
// B (32x16 f16): lane<16 -> col=lane, K {0..7, 16..23}; lane>=16 -> col=lane-16, K {8..15, 24..31}
__device__ __forceinline__ v16h frag_b(const _Float16* base, int col, int kbase, int lane) {
  const _Float16* p = base + (col + (lane & 15)) * LDA + kbase + ((lane >> 4) << 3);
  v8h lo = *(const v8h*)p;
  v8h hi = *(const v8h*)(p + 16);
  return __builtin_shufflevector(lo, hi, 0,1,2,3,4,5,6,7,8,9,10,11,12,13,14,15);
}

// ---------------- weight repack: w[n][ci][kd][kh][kw] f32 -> [tap][n][ci] f16 --
__global__ __launch_bounds__(256) void repack_w(const float* __restrict__ w0,
                                                const float* __restrict__ w1,
                                                const float* __restrict__ w2,
                                                _Float16* __restrict__ out) {
  const int per = NTAPS * WSLICE;
  int i = blockIdx.x * 256 + threadIdx.x;
  if (i >= 3 * per) return;
  int which = i / per, r = i - which * per;
  int tap = r >> 14;            // /16384
  int q   = r & 16383;
  int n   = q >> 7, ci = q & 127;
  const float* w = which == 0 ? w0 : (which == 1 ? w1 : w2);
  out[i] = (_Float16)w[n * (C_CH * NTAPS) + ci * NTAPS + tap];
}

// ---------------- implicit-GEMM 3x3x3 conv via WMMA --------------------------
// one block per (d,h) row: M = 64 voxels (w), N = 128 out ch, K = 27*128.
// B (weights) double-buffered in LDS via GLOBAL_LOAD_ASYNC_TO_LDS_B128 so the
// next tap's 32KB weight tile is DMA'd while the current tap's WMMAs run.
__global__ __launch_bounds__(256) void conv3d_wmma(const float* __restrict__ in,
                                                   const _Float16* __restrict__ wk,
                                                   const float* __restrict__ bias,
                                                   float* __restrict__ out) {
  __shared__ _Float16 sA[64 * LDA];
  __shared__ _Float16 sB[2][128 * LDA];
  int bx = blockIdx.x;
  int d = bx >> 6, h = bx & 63;
  int tid = threadIdx.x;
  int lane = tid & 31, wv = tid >> 5;  // 8 waves
  int mt = wv >> 1;                    // 0..3  (M tile)
  int nt0 = (wv & 1) * 4;              // 0/4   (4 N tiles per wave)
  v8f acc[4] = {};
  int mload = tid & 63;                // w coordinate within row
  int ci0 = tid >> 6;                  // 0..3

  // per-thread async-chunk geometry: 2048 b128 chunks / 256 threads = 8 each
  unsigned ldsOff[8];
#pragma unroll
  for (int k = 0; k < 8; ++k) {
    int c8 = tid + k * 256;
    int n = c8 >> 4, cc = (c8 & 15) << 3;
    ldsOff[k] = (unsigned)((n * LDA + cc) * 2);  // byte offset, 16B aligned
  }
  unsigned bufAddr[2];
  bufAddr[0] = (unsigned)(uintptr_t)(&sB[0][0]);
  bufAddr[1] = (unsigned)(uintptr_t)(&sB[1][0]);

  // preload tap 0 weight tile into buffer 0 (async DMA to LDS)
#pragma unroll
  for (int k = 0; k < 8; ++k) {
    unsigned lo = bufAddr[0] + ldsOff[k];
    const _Float16* g = wk + (tid + k * 256) * 8;
    asm volatile("global_load_async_to_lds_b128 %0, %1, off"
                 :: "v"(lo), "v"(g) : "memory");
  }

#pragma unroll 1
  for (int tap = 0; tap < NTAPS; ++tap) {
    int kd = tap / 9, r9 = tap - kd * 9, kh = r9 / 3, kw = r9 - kh * 3;
    int dd = d + kd - 1, hh = h + kh - 1, ww = mload + kw - 1;
    bool ok = ((unsigned)dd < (unsigned)DVOL) && ((unsigned)hh < (unsigned)HVOL) &&
              ((unsigned)ww < (unsigned)WVOL);
    int base = dd * DSTRIDE + hh * WVOL + ww;
    // A tile: 64 voxels x 128 ci, coalesced along w, f32 -> f16
    for (int ci = ci0; ci < C_CH; ci += 4) {
      float v = 0.f;
      if (ok) v = in[ci * VOXELS + base];
      sA[mload * LDA + ci] = (_Float16)v;
    }
    // kick off async DMA of next tap's weights into the other buffer
    if (tap + 1 < NTAPS) {
      const _Float16* gbase = wk + (tap + 1) * WSLICE;
      unsigned bn = bufAddr[(tap + 1) & 1];
#pragma unroll
      for (int k = 0; k < 8; ++k) {
        unsigned lo = bn + ldsOff[k];
        const _Float16* g = gbase + (tid + k * 256) * 8;
        asm volatile("global_load_async_to_lds_b128 %0, %1, off"
                     :: "v"(lo), "v"(g) : "memory");
      }
      // in-order completion: <=8 outstanding => current tap's 8 chunks landed
      asm volatile("s_wait_asynccnt 0x8" ::: "memory");
    } else {
      asm volatile("s_wait_asynccnt 0x0" ::: "memory");
    }
    __syncthreads();
    const _Float16* sBc = sB[tap & 1];
#pragma unroll
    for (int ks = 0; ks < 4; ++ks) {
      v16h a = frag_a(sA, mt * 16, ks * 32, lane);
#pragma unroll
      for (int j = 0; j < 4; ++j) {
        v16h b = frag_b(sBc, (nt0 + j) * 16, ks * 32, lane);
        acc[j] = __builtin_amdgcn_wmma_f32_16x16x32_f16(false, a, false, b,
                                                        (short)0, acc[j], false, false);
      }
    }
    __syncthreads();  // also fences buffer reuse for the async writes at tap+1
  }
  // C tile store: lane<16 -> N=lane, M=i ; lane>=16 -> N=lane-16, M=i+8 (contig along w)
  int nl = lane & 15, mb = (lane >> 4) * 8;
  int vox = d * DSTRIDE + h * WVOL + mt * 16 + mb;
#pragma unroll
  for (int j = 0; j < 4; ++j) {
    int n = (nt0 + j) * 16 + nl;
    float bv = bias[n];
    float* op = out + n * VOXELS + vox;
#pragma unroll
    for (int i = 0; i < 8; ++i) op[i] = acc[j][i] + bv;
  }
}

// ---------------- attention helpers ------------------------------------------
__device__ __forceinline__ int voxel_of(int win, int t) {
  int bd = win >> 10, bh = (win >> 5) & 31, bw = win & 31;
  int d = bd * 2 + (t >> 2), h = bh * 2 + ((t >> 1) & 1), w = bw * 2 + (t & 1);
  return d * DSTRIDE + h * WVOL + w;
}

// LayerNorm 32 tokens x 128 ch -> f16 tile
__device__ void ln_load(const float* __restrict__ src, const float* __restrict__ g,
                        const float* __restrict__ bsh, _Float16* dst, float* sStat,
                        int win0, int tid) {
  if (tid < 32) {
    int win = win0 + (tid >> 3), t = tid & 7;
    int vx = voxel_of(win, t);
    float s = 0.f, s2 = 0.f;
    for (int c = 0; c < C_CH; ++c) { float v = src[c * VOXELS + vx]; s += v; s2 += v * v; }
    float m = s * (1.f / 128.f);
    float var = s2 * (1.f / 128.f) - m * m;
    sStat[tid * 2] = m;
    sStat[tid * 2 + 1] = rsqrtf(var + 1e-5f);
  }
  __syncthreads();
  for (int idx = tid; idx < 32 * C_CH; idx += 128) {
    int mrow = idx & 31, c = idx >> 5;
    int win = win0 + (mrow >> 3), t = mrow & 7;
    float v = src[c * VOXELS + voxel_of(win, t)];
    float nv = (v - sStat[mrow * 2]) * sStat[mrow * 2 + 1] * g[c] + bsh[c];
    dst[mrow * LDA + c] = (_Float16)nv;
  }
  __syncthreads();
}

// [32x128] @ W^T (W: [n][ci] f32) -> f16 LDS tile, B streamed in 32-row chunks
__device__ void gemm32_lds(const _Float16* sAt, const float* __restrict__ Bg,
                           _Float16* sB, _Float16* sOut, float scale, int tid) {
  int lane = tid & 31, wv = tid >> 5;  // 4 waves
  int mt = wv >> 1, nt = wv & 1;
  for (int nc = 0; nc < 4; ++nc) {
    const float4* bs = (const float4*)(Bg + nc * 32 * C_CH);
    for (int v4 = tid; v4 < 32 * C_CH / 4; v4 += 128) {
      float4 f = bs[v4];
      int n = v4 >> 5, cc = (v4 & 31) * 4;
      _Float16* p = sB + n * LDA + cc;
      p[0] = (_Float16)f.x; p[1] = (_Float16)f.y; p[2] = (_Float16)f.z; p[3] = (_Float16)f.w;
    }
    __syncthreads();
    v8f acc = {};
#pragma unroll
    for (int ks = 0; ks < 4; ++ks) {
      v16h a = frag_a(sAt, mt * 16, ks * 32, lane);
      v16h b = frag_b(sB, nt * 16, ks * 32, lane);
      acc = __builtin_amdgcn_wmma_f32_16x16x32_f16(false, a, false, b, (short)0, acc,
                                                   false, false);
    }
    int nl = lane & 15, mb = (lane >> 4) * 8;
    int n = nc * 32 + nt * 16 + nl;
#pragma unroll
    for (int i = 0; i < 8; ++i)
      sOut[(mt * 16 + mb + i) * LDA + n] = (_Float16)(acc[i] * scale);
    __syncthreads();
  }
}

// proj: [32x128] @ proj_w^T + b -> scatter f32 to (C,D,H,W) volume
__device__ void gemm32_proj(const _Float16* sAt, const float* __restrict__ Bg,
                            const float* __restrict__ bias, _Float16* sB,
                            float* __restrict__ out, int win0, int tid) {
  int lane = tid & 31, wv = tid >> 5;
  int mt = wv >> 1, nt = wv & 1;
  for (int nc = 0; nc < 4; ++nc) {
    const float4* bs = (const float4*)(Bg + nc * 32 * C_CH);
    for (int v4 = tid; v4 < 32 * C_CH / 4; v4 += 128) {
      float4 f = bs[v4];
      int n = v4 >> 5, cc = (v4 & 31) * 4;
      _Float16* p = sB + n * LDA + cc;
      p[0] = (_Float16)f.x; p[1] = (_Float16)f.y; p[2] = (_Float16)f.z; p[3] = (_Float16)f.w;
    }
    __syncthreads();
    v8f acc = {};
#pragma unroll
    for (int ks = 0; ks < 4; ++ks) {
      v16h a = frag_a(sAt, mt * 16, ks * 32, lane);
      v16h b = frag_b(sB, nt * 16, ks * 32, lane);
      acc = __builtin_amdgcn_wmma_f32_16x16x32_f16(false, a, false, b, (short)0, acc,
                                                   false, false);
    }
    int nl = lane & 15, mb = (lane >> 4) * 8;
    int n = nc * 32 + nt * 16 + nl;
    float bv = bias[n];
#pragma unroll
    for (int i = 0; i < 8; ++i) {
      int mrow = mt * 16 + mb + i;
      int win = win0 + (mrow >> 3), t = mrow & 7;
      out[n * VOXELS + voxel_of(win, t)] = acc[i] + bv;
    }
    __syncthreads();
  }
}

// ---------------- windowed MHCA: 4 windows (32 tokens) per 128-thread block ---
__global__ __launch_bounds__(128) void attn_kernel(
    const float* __restrict__ q_enc, const float* __restrict__ k_enc,
    const float* __restrict__ norm_g, const float* __restrict__ norm_b,
    const float* __restrict__ q_w, const float* __restrict__ k_w,
    const float* __restrict__ v_w,
    const float* __restrict__ pre_aw1, const float* __restrict__ pre_aw2,
    const float* __restrict__ pre_mw1, const float* __restrict__ pre_mw2,
    const float* __restrict__ proj_w, const float* __restrict__ proj_b,
    float* __restrict__ xo) {
  __shared__ _Float16 sT0[32 * LDA];   // LN(q), later V
  __shared__ _Float16 sT1[32 * LDA];   // LN(k), later XO
  __shared__ _Float16 sQ[32 * LDA];
  __shared__ _Float16 sKt[32 * LDA];
  __shared__ _Float16 sB[32 * LDA];
  __shared__ float sLog[4 * 8 * 8 * 8];  // win,head,q,k
  __shared__ float sStat[32 * 2];
  __shared__ float sAvg[32], sMx[32], sGate[32];

  int tid = threadIdx.x;
  int win0 = blockIdx.x * 4;

  ln_load(q_enc, norm_g, norm_b, sT0, sStat, win0, tid);
  ln_load(k_enc, norm_g, norm_b, sT1, sStat, win0, tid);

  gemm32_lds(sT0, q_w, sB, sQ, 0.25f, tid);   // Q * head_dim^-0.5
  gemm32_lds(sT1, k_w, sB, sKt, 1.0f, tid);
  gemm32_lds(sT1, v_w, sB, sT0, 1.0f, tid);   // V overwrites LN(q)

  // logits: per (win,head) 8x8, dot over head_dim=16
  {
    int u = tid >> 2, sub = tid & 3;  // u<32: (win,head)
    int wi = u >> 3, hh = u & 7;
    for (int qr = sub * 2; qr < sub * 2 + 2; ++qr) {
      int qm = wi * 8 + qr;
      for (int kc = 0; kc < 8; ++kc) {
        int km = wi * 8 + kc;
        float a = 0.f;
#pragma unroll
        for (int dd = 0; dd < 16; ++dd)
          a += (float)sQ[qm * LDA + hh * 16 + dd] * (float)sKt[km * LDA + hh * 16 + dd];
        sLog[(u * 8 + qr) * 8 + kc] = a;
      }
    }
  }
  __syncthreads();
  // per-(win,head) mean/max over the 8x8 map
  if (tid < 32) {
    float s = 0.f, mx = -1e30f;
    const float* lp = sLog + tid * 64;
    for (int i = 0; i < 64; ++i) { s += lp[i]; mx = fmaxf(mx, lp[i]); }
    sAvg[tid] = s * (1.f / 64.f);
    sMx[tid] = mx;
  }
  __syncthreads();
  // pre-attn SE gate: heads-channel MLP 8->2->8
  if (tid < 32) {
    int wi = tid >> 3, hh = tid & 7;
    const float* av = sAvg + wi * 8;
    const float* mv = sMx + wi * 8;
    float ha0 = 0, ha1 = 0, hm0 = 0, hm1 = 0;
    for (int c = 0; c < 8; ++c) {
      ha0 += av[c] * pre_aw1[c];     ha1 += av[c] * pre_aw1[8 + c];
      hm0 += mv[c] * pre_mw1[c];     hm1 += mv[c] * pre_mw1[8 + c];
    }
    ha0 = fmaxf(ha0, 0.f); ha1 = fmaxf(ha1, 0.f);
    hm0 = fmaxf(hm0, 0.f); hm1 = fmaxf(hm1, 0.f);
    float g = pre_aw2[hh * 2] * ha0 + pre_aw2[hh * 2 + 1] * ha1 +
              pre_mw2[hh * 2] * hm0 + pre_mw2[hh * 2 + 1] * hm1;
    sGate[tid] = 1.f / (1.f + __expf(-g));
  }
  __syncthreads();
  // gate + softmax over k (rows of 8)
  for (int r = tid; r < 256; r += 128) {
    float* row = sLog + r * 8;
    float g = sGate[r >> 3];
    float mx = -1e30f;
    for (int k = 0; k < 8; ++k) { row[k] *= g; mx = fmaxf(mx, row[k]); }
    float s = 0.f;
    for (int k = 0; k < 8; ++k) { float e = __expf(row[k] - mx); row[k] = e; s += e; }
    float inv = 1.f / s;
    for (int k = 0; k < 8; ++k) row[k] *= inv;
  }
  __syncthreads();
  // attn @ V -> XO f16 tile (overwrites LN(k))
  {
    int u = tid >> 2, sub = tid & 3;
    int wi = u >> 3, hh = u & 7;
    for (int qr = sub * 2; qr < sub * 2 + 2; ++qr) {
      const float* arow = sLog + (u * 8 + qr) * 8;
      int qm = wi * 8 + qr;
      for (int dd = 0; dd < 16; ++dd) {
        float a = 0.f;
#pragma unroll
        for (int k = 0; k < 8; ++k)
          a += arow[k] * (float)sT0[(wi * 8 + k) * LDA + hh * 16 + dd];
        sT1[qm * LDA + hh * 16 + dd] = (_Float16)a;
      }
    }
  }
  __syncthreads();
  gemm32_proj(sT1, proj_w, proj_b, sB, xo, win0, tid);
}

// ---------------- post-attn SE over full volume -------------------------------
__global__ __launch_bounds__(256) void reduce_cm(const float* __restrict__ x,
                                                 float* __restrict__ avg,
                                                 float* __restrict__ mx) {
  __shared__ float ss[256], sm[256];
  int c = blockIdx.x;
  const float* p = x + (size_t)c * VOXELS;
  float s = 0.f, m = -1e30f;
  for (int i = threadIdx.x; i < VOXELS; i += 256) { float v = p[i]; s += v; m = fmaxf(m, v); }
  ss[threadIdx.x] = s; sm[threadIdx.x] = m;
  __syncthreads();
  for (int st = 128; st > 0; st >>= 1) {
    if (threadIdx.x < (unsigned)st) {
      ss[threadIdx.x] += ss[threadIdx.x + st];
      sm[threadIdx.x] = fmaxf(sm[threadIdx.x], sm[threadIdx.x + st]);
    }
    __syncthreads();
  }
  if (threadIdx.x == 0) { avg[c] = ss[0] * (1.f / (float)VOXELS); mx[c] = sm[0]; }
}

__global__ __launch_bounds__(128) void se_gate(const float* __restrict__ avg,
                                               const float* __restrict__ mx,
                                               const float* __restrict__ aw1,
                                               const float* __restrict__ aw2,
                                               const float* __restrict__ mw1,
                                               const float* __restrict__ mw2,
                                               float* __restrict__ gate) {
  __shared__ float ha[32], hm[32];
  int tid = threadIdx.x;
  if (tid < 32) {
    float a = 0.f, m = 0.f;
    for (int c = 0; c < C_CH; ++c) {
      a += avg[c] * aw1[tid * C_CH + c];
      m += mx[c] * mw1[tid * C_CH + c];
    }
    ha[tid] = fmaxf(a, 0.f);
    hm[tid] = fmaxf(m, 0.f);
  }
  __syncthreads();
  float g = 0.f;
  for (int j = 0; j < 32; ++j) g += ha[j] * aw2[tid * 32 + j] + hm[j] * mw2[tid * 32 + j];
  gate[tid] = 1.f / (1.f + __expf(-g));
}

__global__ __launch_bounds__(256) void fuse_gate_skip(float* __restrict__ xo,
                                                      const float* __restrict__ skip,
                                                      const float* __restrict__ gate) {
  int i = blockIdx.x * 256 + threadIdx.x;
  float g = gate[i >> 17];   // VOXELS == 1<<17
  xo[i] = xo[i] * g + skip[i];
}

// ---------------- orchestration -----------------------------------------------
extern "C" void kernel_launch(void* const* d_in, const int* in_sizes, int n_in,
                              void* d_out, int out_size, void* d_ws, size_t ws_size,
                              hipStream_t stream) {
  (void)in_sizes; (void)n_in; (void)out_size; (void)ws_size;
  const float* x_q      = (const float*)d_in[0];
  const float* x_k      = (const float*)d_in[1];
  const float* conv_q_w = (const float*)d_in[2];
  const float* conv_q_b = (const float*)d_in[3];
  const float* conv_k_w = (const float*)d_in[4];
  const float* conv_k_b = (const float*)d_in[5];
  const float* norm_g   = (const float*)d_in[6];
  const float* norm_b   = (const float*)d_in[7];
  const float* q_w      = (const float*)d_in[8];
  const float* k_w      = (const float*)d_in[9];
  const float* v_w      = (const float*)d_in[10];
  const float* pre_aw1  = (const float*)d_in[11];
  const float* pre_aw2  = (const float*)d_in[12];
  const float* pre_mw1  = (const float*)d_in[13];
  const float* pre_mw2  = (const float*)d_in[14];
  const float* proj_w   = (const float*)d_in[15];
  const float* proj_b   = (const float*)d_in[16];
  const float* post_aw1 = (const float*)d_in[17];
  const float* post_aw2 = (const float*)d_in[18];
  const float* post_mw1 = (const float*)d_in[19];
  const float* post_mw2 = (const float*)d_in[20];
  const float* fusion_w = (const float*)d_in[21];
  const float* fusion_b = (const float*)d_in[22];

  const size_t vol = (size_t)C_CH * VOXELS;  // 16777216 floats
  float* q_enc = (float*)d_ws;
  float* k_enc = q_enc + vol;
  float* xo    = k_enc + vol;
  _Float16* wpk = (_Float16*)(xo + vol);     // 3 * 27 * 16384 halves
  float* avgc  = (float*)(wpk + (size_t)3 * NTAPS * WSLICE);
  float* maxc  = avgc + C_CH;
  float* gate2 = maxc + C_CH;

  repack_w<<<(3 * NTAPS * WSLICE + 255) / 256, 256, 0, stream>>>(conv_q_w, conv_k_w,
                                                                 fusion_w, wpk);
  conv3d_wmma<<<2048, 256, 0, stream>>>(x_q, wpk, conv_q_b, q_enc);
  conv3d_wmma<<<2048, 256, 0, stream>>>(x_k, wpk + NTAPS * WSLICE, conv_k_b, k_enc);
  attn_kernel<<<4096, 128, 0, stream>>>(q_enc, k_enc, norm_g, norm_b, q_w, k_w, v_w,
                                        pre_aw1, pre_aw2, pre_mw1, pre_mw2,
                                        proj_w, proj_b, xo);
  reduce_cm<<<C_CH, 256, 0, stream>>>(xo, avgc, maxc);
  se_gate<<<1, 128, 0, stream>>>(avgc, maxc, post_aw1, post_aw2, post_mw1, post_mw2,
                                 gate2);
  fuse_gate_skip<<<(unsigned)(vol / 256), 256, 0, stream>>>(xo, q_enc, gate2);
  conv3d_wmma<<<2048, 256, 0, stream>>>(xo, wpk + 2 * NTAPS * WSLICE, fusion_b,
                                        (float*)d_out);
}
